// CMAT_4561255269047
// MI455X (gfx1250) — matrix operand order repaired and checked by reading, not compiled, not run on hardware
//
#include <hip/hip_runtime.h>
#include <hip/hip_bf16.h>

typedef _Float16 half_t;
typedef __attribute__((ext_vector_type(8)))  _Float16 h8;    // 16B
typedef __attribute__((ext_vector_type(16))) _Float16 v16h;
typedef __attribute__((ext_vector_type(8)))  float    v8f;

#define BM 64
#define BN 128
#define BK 32
#define PADK 40     // BK + 8 halves -> 80B rows (16B aligned), kills bank conflicts
#define NPAD 2048   // padded pixel/N stride (1936 -> 2048)

// ---- CDNA5 async global->LDS (ASYNCcnt-tracked DMA; bypasses VGPRs) ----------
__device__ __forceinline__ void async_ld_b128(unsigned lds_off, const void* gaddr)
{
    asm volatile("global_load_async_to_lds_b128 %0, %1, off"
                 :: "v"(lds_off), "v"(gaddr) : "memory");
}
__device__ __forceinline__ void wait_async0()
{
    asm volatile("s_wait_asynccnt 0x0" ::: "memory");
}

// ---------------------------------------------------------------------------
// WMMA GEMM on PADDED operands:  M%64==0, N%128==0, K%32==0. No bounds checks.
//   C[M][N](strided) = A[M][K](aStride) x B[K][N](bStride) * scale + shift, relu
// CONV=1: B generated on the fly (3x3 SAME im2col) from f16 planes with channel
//         stride xcs; pixels >= Nreal produce zeros.
// CONV=0: dense f16 B (K-pad rows must be zero if matching A columns are not).
// ---------------------------------------------------------------------------
template <int CONV>
__global__ __launch_bounds__(256) void gemm_wmma_kernel(
    const half_t* __restrict__ A, const half_t* __restrict__ Bsrc,
    long aBS, long bBS,
    int M, int N, int K,
    int aStride, int bStride,
    int Nreal, int xcs, int H, int W,
    const float* __restrict__ scale, const float* __restrict__ shift, int relu,
    float* __restrict__ Cf32, long c32BS, int c32Stride,
    half_t* __restrict__ Cf16, long c16BS, int c16Stride, int transF16)
{
    __shared__ half_t Alds[BM][PADK];   // [m][k]
    __shared__ half_t Blds[BN][PADK];   // transposed: [n][k]

    const int m0 = blockIdx.x * BM;
    const int n0 = blockIdx.y * BN;
    const int b  = blockIdx.z;
    A += (long)b * aBS;
    const half_t* Bp = Bsrc + (long)b * bBS;
    if (Cf32) Cf32 += (long)b * c32BS;
    if (Cf16) Cf16 += (long)b * c16BS;

    const int tid  = threadIdx.x;
    const int lane = tid & 31;
    const int wid  = tid >> 5;
    const int wm   = (wid >> 2) * 32;
    const int wn   = (wid & 3) * 32;
    const int l15  = lane & 15;
    const int kh   = lane >> 4;

    // per-thread staging coordinates (loop-invariant)
    const int arow = tid >> 2;            // 0..63
    const int akc  = (tid & 3) * 8;       // 0,8,16,24
    const half_t*  aSrc   = A + (long)(m0 + arow) * aStride + akc;
    const unsigned aLds   = (unsigned)(uintptr_t)&Alds[arow][akc]; // LDS offset

    v8f acc[2][2] = {};

    for (int k0 = 0; k0 < K; k0 += BK) {
        // ---- A tile: async DMA global->LDS (16B per lane), overlaps B staging
        async_ld_b128(aLds, aSrc + k0);
        if (k0 + BK < K)
            __builtin_prefetch(aSrc + k0 + BK, 0, 3);

        // ---- B tile (transposed into LDS)
#pragma unroll
        for (int it = 0; it < 2; ++it) {
            int q  = tid + it * 256;      // 0..511
            int kl = q >> 4;              // 0..31
            int nb = (q & 15) * 8;        // 0..120
            int kg = k0 + kl;
            h8 vals;
            if (CONV) {
                int ci  = kg / 9;
                int tap = kg - ci * 9;
                int dy  = tap / 3 - 1;
                int dx  = tap - (tap / 3) * 3 - 1;
                const half_t* xp = Bp + (long)ci * xcs;
                int ng = n0 + nb;
                int h  = ng / W;
                int w  = ng - h * W;
#pragma unroll
                for (int j = 0; j < 8; ++j) {
                    int h2 = h + dy, w2 = w + dx;
                    bool ok = (ng < Nreal) & ((unsigned)h2 < (unsigned)H) &
                              ((unsigned)w2 < (unsigned)W);
                    int h2c = min(max(h2, 0), H - 1);   // clamped addr: load is
                    int w2c = min(max(w2, 0), W - 1);   // unconditional
                    half_t t = xp[h2c * W + w2c];
                    vals[j] = ok ? t : (half_t)0.f;
                    ++ng; ++w;
                    if (w == W) { w = 0; ++h; }
                }
            } else {
                vals = *(const h8*)(Bp + (long)kg * bStride + n0 + nb);
            }
#pragma unroll
            for (int j = 0; j < 8; ++j) Blds[nb + j][kl] = vals[j];
        }
        wait_async0();          // A-tile DMA landed in LDS
        __syncthreads();

        // ---- fragments (contiguous k order, same in A and B):
        // A lane: halves j   -> K=kh*16+j ; M=l15
        // B lane: halves j   -> K=kh*16+j ; N=l15
        v16h af[2], bf[2];
#pragma unroll
        for (int f = 0; f < 2; ++f) {
            const half_t* ar = &Alds[wm + f * 16 + l15][0];
#pragma unroll
            for (int j = 0; j < 8; ++j) {
                af[f][j]     = ar[kh * 16 + j];
                af[f][8 + j] = ar[kh * 16 + 8 + j];
            }
            const half_t* br = &Blds[wn + f * 16 + l15][kh * 16];
#pragma unroll
            for (int j = 0; j < 16; ++j) bf[f][j] = br[j];
        }
#pragma unroll
        for (int fm = 0; fm < 2; ++fm)
#pragma unroll
            for (int fn = 0; fn < 2; ++fn)
                acc[fm][fn] = __builtin_amdgcn_wmma_f32_16x16x32_f16(
                    false, af[fm], false, bf[fn], (short)0, acc[fm][fn],
                    false, false);
        __syncthreads();
    }

    // ---- epilogue (C lane layout: VGPR r -> M=kh*8+r, N=l15) — unconditional
#pragma unroll
    for (int fm = 0; fm < 2; ++fm)
        for (int fn = 0; fn < 2; ++fn)
            for (int r = 0; r < 8; ++r) {
                int gm = m0 + wm + fm * 16 + kh * 8 + r;
                int gn = n0 + wn + fn * 16 + l15;
                float v = acc[fm][fn][r];
                if (scale) v *= scale[gm];
                if (shift) v += shift[gm];
                if (relu)  v  = fmaxf(v, 0.f);
                if (Cf32)  Cf32[(long)gm * c32Stride + gn] = v;
                if (Cf16) {
                    long idx = transF16 ? ((long)gn * c16Stride + gm)
                                        : ((long)gm * c16Stride + gn);
                    Cf16[idx] = (half_t)v;
                }
            }
}

// ---------------------------------------------------------------------------
__global__ void cvt_f32_f16(const float* __restrict__ s, half_t* __restrict__ d, long n)
{
    long i = (long)blockIdx.x * 256 + threadIdx.x;
    if (i < n) d[i] = (half_t)s[i];
}

__global__ void zero_f16(half_t* __restrict__ d, long n)
{
    long i = (long)blockIdx.x * 256 + threadIdx.x;
    if (i < n) d[i] = (half_t)0.f;
}

// out[i] = i < n ? in[i] : 0   (bias zero-padding to Mpad)
__global__ void pad_bias_kernel(const float* __restrict__ in, float* __restrict__ out,
                                int n, int npad)
{
    int i = blockIdx.x * 256 + threadIdx.x;
    if (i < npad) out[i] = (i < n) ? in[i] : 0.f;
}

// scale = g / sqrt(v+eps);  shift = (conv_bias - mean)*scale + beta
__global__ void bn_coeff_kernel(const float* g, const float* bb, const float* m,
                                const float* v, const float* cb,
                                float* scale, float* shift, int C)
{
    int i = blockIdx.x * 256 + threadIdx.x;
    if (i < C) {
        float s  = g[i] * rsqrtf(v[i] + 1e-5f);
        scale[i] = s;
        shift[i] = (cb[i] - m[i]) * s + bb[i];
    }
}

// r = relu(o2[:C]*o1 + o2[C:2C]); o1/o2/r16 use NPAD row stride, r32 is flat
__global__ void fuse_sa_kernel(const half_t* __restrict__ o2,
                               const half_t* __restrict__ o1,
                               float* __restrict__ r32, half_t* __restrict__ r16,
                               int C, long NP, long n)
{
    long i = (long)blockIdx.x * 256 + threadIdx.x;
    if (i >= n) return;
    long b   = i / ((long)C * NP);
    long rem = i - b * (long)C * NP;
    long c   = rem / NP;
    long p   = rem - c * NP;
    float w_ = (float)o2[((b * 2 * C) + c) * NPAD + p];
    float b_ = (float)o2[((b * 2 * C) + C + c) * NPAD + p];
    float v  = fmaxf(w_ * (float)o1[(b * C + c) * NPAD + p] + b_, 0.f);
    r32[i] = v;
    r16[(b * C + c) * NPAD + p] = (half_t)v;
}

// softmax over row i of S[b][i][:NP]; writes transposed PT[b][j][i] (stride NPAD)
// and zeroes the K-pad rows NP..Kpad-1 of column i.
__global__ void softmax_rows_kernel(const float* __restrict__ S,
                                    half_t* __restrict__ PT, int NP, int Kpad)
{
    int i = blockIdx.x;
    int b = blockIdx.y;
    const float* row = S + ((long)b * NPAD + i) * NPAD;
    __shared__ float red[256];
    float mx = -1e30f;
    for (int j = threadIdx.x; j < NP; j += 256) mx = fmaxf(mx, row[j]);
    red[threadIdx.x] = mx; __syncthreads();
    for (int s = 128; s > 0; s >>= 1) {
        if (threadIdx.x < s) red[threadIdx.x] = fmaxf(red[threadIdx.x], red[threadIdx.x + s]);
        __syncthreads();
    }
    mx = red[0]; __syncthreads();
    float sum = 0.f;
    for (int j = threadIdx.x; j < NP; j += 256) sum += __expf(row[j] - mx);
    red[threadIdx.x] = sum; __syncthreads();
    for (int s = 128; s > 0; s >>= 1) {
        if (threadIdx.x < s) red[threadIdx.x] += red[threadIdx.x + s];
        __syncthreads();
    }
    float inv = 1.f / red[0];
    half_t* pc = PT + (long)b * NPAD * NPAD + i;
    for (int j = threadIdx.x; j < NP; j += 256)
        pc[(long)j * NPAD] = (half_t)(__expf(row[j] - mx) * inv);
    for (int j = NP + threadIdx.x; j < Kpad; j += 256)
        pc[(long)j * NPAD] = (half_t)0.f;
}

__global__ void combine_kernel(const float* __restrict__ r32, const float* __restrict__ d32,
                               const float* __restrict__ f1,  const float* __restrict__ f2,
                               const float* beta, const float* gamma, const float* gate,
                               float* __restrict__ out, int C, long NP, long n)
{
    long i = (long)blockIdx.x * 256 + threadIdx.x;
    if (i >= n) return;
    long b   = i / ((long)C * NP);
    long rem = i - b * (long)C * NP;
    long c   = rem / NP;
    long p   = rem - c * NP;
    long fi  = (b * C + c) * NPAD + p;     // feat buffers use NPAD stride
    float g  = *gate;
    out[i]     = r32[i] + g * (*beta) * f1[fi];
    out[n + i] = d32[i] + (1.f - g) * (*gamma) * f2[fi];
}

// ---------------------------------------------------------------------------
extern "C" void kernel_launch(void* const* d_in, const int* in_sizes, int n_in,
                              void* d_out, int out_size, void* d_ws, size_t ws_size,
                              hipStream_t stream)
{
    (void)in_sizes; (void)n_in; (void)out_size; (void)ws_size;

    const long Bb = 8, Cin = 512, C = 256, Cq = 32, H = 44, W = 44;
    const long NP = H * W;          // 1936
    const long K1 = Cin * 9;        // 4608 (mult of 32)
    const long K2 = C * 9;          // 2304 (mult of 32)
    const long CqP = 64;            // q/k rows padded 32 -> 64
    const long MS  = 1984;          // score rows padded 1936 -> 1984 (mult 64)
    const long KF  = 1952;          // feat K padded 1936 -> 1952 (mult 32)

    const float* rgb   = (const float*)d_in[0];
    const float* depth = (const float*)d_in[1];
    const float* beta  = (const float*)d_in[2];
    const float* gamma = (const float*)d_in[3];
    const float* gate  = (const float*)d_in[4];
    const float* s1c1w = (const float*)d_in[5];  const float* s1c1b = (const float*)d_in[6];
    const float* s1bg  = (const float*)d_in[7];  const float* s1bb  = (const float*)d_in[8];
    const float* s1bm  = (const float*)d_in[9];  const float* s1bv  = (const float*)d_in[10];
    const float* s1c2w = (const float*)d_in[11]; const float* s1c2b = (const float*)d_in[12];
    const float* s2c1w = (const float*)d_in[13]; const float* s2c1b = (const float*)d_in[14];
    const float* s2bg  = (const float*)d_in[15]; const float* s2bb  = (const float*)d_in[16];
    const float* s2bm  = (const float*)d_in[17]; const float* s2bv  = (const float*)d_in[18];
    const float* s2c2w = (const float*)d_in[19]; const float* s2c2b = (const float*)d_in[20];
    const float* a1qw = (const float*)d_in[21]; const float* a1qb = (const float*)d_in[22];
    const float* a1kw = (const float*)d_in[23]; const float* a1kb = (const float*)d_in[24];
    const float* a1vw = (const float*)d_in[25]; const float* a1vb = (const float*)d_in[26];
    const float* a2qw = (const float*)d_in[27]; const float* a2qb = (const float*)d_in[28];
    const float* a2kw = (const float*)d_in[29]; const float* a2kb = (const float*)d_in[30];
    const float* a2vw = (const float*)d_in[31]; const float* a2vb = (const float*)d_in[32];

    // ---- carve workspace (256B aligned)
    char* base = (char*)d_ws;
    size_t off = 0;
    auto carve = [&](size_t bytes) -> void* {
        void* p = base + off;
        off = (off + bytes + 255) & ~(size_t)255;
        return p;
    };
    half_t* xh_r = (half_t*)carve(Bb * Cin * NP * 2);       // flat NCHW f16
    half_t* xh_d = (half_t*)carve(Bb * Cin * NP * 2);
    half_t* w1a  = (half_t*)carve(C * K1 * 2);
    half_t* w1b  = (half_t*)carve(2 * C * K2 * 2);
    half_t* w2a  = (half_t*)carve(C * K1 * 2);
    half_t* w2b  = (half_t*)carve(2 * C * K2 * 2);
    half_t* wq1  = (half_t*)carve(CqP * C * 2);             // zero-padded to 64 rows
    half_t* wk1  = (half_t*)carve(CqP * C * 2);
    half_t* wv1  = (half_t*)carve(C * C * 2);
    half_t* wq2  = (half_t*)carve(CqP * C * 2);
    half_t* wk2  = (half_t*)carve(CqP * C * 2);
    half_t* wv2  = (half_t*)carve(C * C * 2);
    float*  qb1p = (float*)carve(CqP * 4);                  // zero-padded biases
    float*  kb1p = (float*)carve(CqP * 4);
    float*  qb2p = (float*)carve(CqP * 4);
    float*  kb2p = (float*)carve(CqP * 4);
    float*  sc1  = (float*)carve(C * 4);
    float*  sh1  = (float*)carve(C * 4);
    float*  sc2  = (float*)carve(C * 4);
    float*  sh2  = (float*)carve(C * 4);
    half_t* o1r  = (half_t*)carve(Bb * C * NPAD * 2);       // padded [C][NPAD]
    half_t* o1d  = (half_t*)carve(Bb * C * NPAD * 2);
    half_t* o2r  = (half_t*)carve(Bb * 2 * C * NPAD * 2);
    half_t* o2d  = (half_t*)carve(Bb * 2 * C * NPAD * 2);
    float*  r32  = (float*)carve(Bb * C * NP * 4);          // flat
    float*  d32  = (float*)carve(Bb * C * NP * 4);
    half_t* r16  = (half_t*)carve(Bb * C * NPAD * 2);
    half_t* d16  = (half_t*)carve(Bb * C * NPAD * 2);
    half_t* qT   = (half_t*)carve(Bb * NPAD * CqP * 2);     // [n][64]
    half_t* kB   = (half_t*)carve(Bb * CqP * NPAD * 2);     // [64][NPAD]
    half_t* vB   = (half_t*)carve(Bb * C * NPAD * 2);
    float*  Sbuf = (float*)carve(Bb * NPAD * NPAD * 4);
    half_t* PT   = (half_t*)carve(Bb * NPAD * NPAD * 2);
    float*  f1   = (float*)carve(Bb * C * NPAD * 4);
    float*  f2   = (float*)carve(Bb * C * NPAD * 4);

    dim3 blk(256);
    auto lin = [](long n) { return dim3((unsigned)((n + 255) / 256)); };
    auto cvt = [&](const float* s, half_t* d, long n) {
        cvt_f32_f16<<<lin(n), dim3(256), 0, stream>>>(s, d, n);
    };

    // pack activations + weights to f16 (OIHW flat == A[m][k], k = ci*9+tap)
    cvt(rgb,   xh_r, Bb * Cin * NP);
    cvt(depth, xh_d, Bb * Cin * NP);
    cvt(s1c1w, w1a, C * K1);  cvt(s1c2w, w1b, 2 * C * K2);
    cvt(s2c1w, w2a, C * K1);  cvt(s2c2w, w2b, 2 * C * K2);
    zero_f16<<<lin(CqP * C), blk, 0, stream>>>(wq1, CqP * C);
    zero_f16<<<lin(CqP * C), blk, 0, stream>>>(wk1, CqP * C);
    zero_f16<<<lin(CqP * C), blk, 0, stream>>>(wq2, CqP * C);
    zero_f16<<<lin(CqP * C), blk, 0, stream>>>(wk2, CqP * C);
    cvt(a1qw, wq1, Cq * C); cvt(a1kw, wk1, Cq * C); cvt(a1vw, wv1, C * C);
    cvt(a2qw, wq2, Cq * C); cvt(a2kw, wk2, Cq * C); cvt(a2vw, wv2, C * C);
    pad_bias_kernel<<<1, 256, 0, stream>>>(a1qb, qb1p, (int)Cq, (int)CqP);
    pad_bias_kernel<<<1, 256, 0, stream>>>(a1kb, kb1p, (int)Cq, (int)CqP);
    pad_bias_kernel<<<1, 256, 0, stream>>>(a2qb, qb2p, (int)Cq, (int)CqP);
    pad_bias_kernel<<<1, 256, 0, stream>>>(a2kb, kb2p, (int)Cq, (int)CqP);
    bn_coeff_kernel<<<1, 256, 0, stream>>>(s1bg, s1bb, s1bm, s1bv, s1c1b, sc1, sh1, (int)C);
    bn_coeff_kernel<<<1, 256, 0, stream>>>(s2bg, s2bb, s2bm, s2bv, s2c1b, sc2, sh2, (int)C);

    auto gdim = [&](long M, long N) {
        return dim3((unsigned)(M / BM), (unsigned)(N / BN), (unsigned)Bb);
    };

    // ---- sa_block(rgb)
    gemm_wmma_kernel<1><<<gdim(C, NPAD), blk, 0, stream>>>(
        w1a, xh_r, 0, Cin * NP, (int)C, NPAD, (int)K1, (int)K1, 0,
        (int)NP, (int)NP, 44, 44, sc1, sh1, 1,
        nullptr, 0, 0, o1r, C * NPAD, NPAD, 0);
    gemm_wmma_kernel<1><<<gdim(2 * C, NPAD), blk, 0, stream>>>(
        w1b, o1r, 0, C * NPAD, (int)(2 * C), NPAD, (int)K2, (int)K2, 0,
        (int)NP, NPAD, 44, 44, nullptr, s1c2b, 0,
        nullptr, 0, 0, o2r, 2 * C * NPAD, NPAD, 0);
    fuse_sa_kernel<<<lin(Bb * C * NP), blk, 0, stream>>>(
        o2r, o1r, r32, r16, (int)C, NP, Bb * C * NP);
    // ---- sa_block(depth)
    gemm_wmma_kernel<1><<<gdim(C, NPAD), blk, 0, stream>>>(
        w2a, xh_d, 0, Cin * NP, (int)C, NPAD, (int)K1, (int)K1, 0,
        (int)NP, (int)NP, 44, 44, sc2, sh2, 1,
        nullptr, 0, 0, o1d, C * NPAD, NPAD, 0);
    gemm_wmma_kernel<1><<<gdim(2 * C, NPAD), blk, 0, stream>>>(
        w2b, o1d, 0, C * NPAD, (int)(2 * C), NPAD, (int)K2, (int)K2, 0,
        (int)NP, NPAD, 44, 44, nullptr, s2c2b, 0,
        nullptr, 0, 0, o2d, 2 * C * NPAD, NPAD, 0);
    fuse_sa_kernel<<<lin(Bb * C * NP), blk, 0, stream>>>(
        o2d, o1d, d32, d16, (int)C, NP, Bb * C * NP);

    // ---- cross-attention (q,k from X; v from Y)
    auto attention = [&](const half_t* wq, const float* qbp, const half_t* wk,
                         const float* kbp, const half_t* wv, const float* vb,
                         const half_t* X, const half_t* Y, float* feat) {
        // q -> transposed qT[n][64]
        gemm_wmma_kernel<0><<<gdim(CqP, NPAD), blk, 0, stream>>>(
            wq, X, 0, C * NPAD, (int)CqP, NPAD, (int)C, (int)C, NPAD,
            0, 0, 0, 0, nullptr, qbp, 0,
            nullptr, 0, 0, qT, NPAD * CqP, (int)CqP, 1);
        // k -> kB[64][NPAD]
        gemm_wmma_kernel<0><<<gdim(CqP, NPAD), blk, 0, stream>>>(
            wk, X, 0, C * NPAD, (int)CqP, NPAD, (int)C, (int)C, NPAD,
            0, 0, 0, 0, nullptr, kbp, 0,
            nullptr, 0, 0, kB, CqP * NPAD, NPAD, 0);
        // v -> vB[C][NPAD]
        gemm_wmma_kernel<0><<<gdim(C, NPAD), blk, 0, stream>>>(
            wv, Y, 0, C * NPAD, (int)C, NPAD, (int)C, (int)C, NPAD,
            0, 0, 0, 0, nullptr, vb, 0,
            nullptr, 0, 0, vB, C * NPAD, NPAD, 0);
        // S[n][m] = qT x kB  (K = 64, pad cols/rows are exact zeros)
        gemm_wmma_kernel<0><<<gdim(MS, NPAD), blk, 0, stream>>>(
            qT, kB, NPAD * CqP, CqP * NPAD, (int)MS, NPAD, (int)CqP,
            (int)CqP, NPAD, 0, 0, 0, 0, nullptr, nullptr, 0,
            Sbuf, (long)NPAD * NPAD, NPAD, nullptr, 0, 0, 0);
        // softmax rows; write PT transposed + zero K-pad rows
        softmax_rows_kernel<<<dim3((unsigned)NP, (unsigned)Bb), blk, 0, stream>>>(
            Sbuf, PT, (int)NP, (int)KF);
        // feat = vB x PT  (K = 1952; PT pad rows zero)
        gemm_wmma_kernel<0><<<gdim(C, NPAD), blk, 0, stream>>>(
            vB, PT, C * NPAD, (long)NPAD * NPAD, (int)C, NPAD, (int)KF,
            NPAD, NPAD, 0, 0, 0, 0, nullptr, nullptr, 0,
            feat, C * NPAD, NPAD, nullptr, 0, 0, 0);
    };

    attention(wq1, qb1p, wk1, kb1p, wv1, a1vb, d16, r16, f1);
    attention(wq2, qb2p, wk2, kb2p, wv2, a2vb, r16, d16, f2);

    combine_kernel<<<lin(Bb * C * NP), blk, 0, stream>>>(
        r32, d32, f1, f2, beta, gamma, gate, (float*)d_out,
        (int)C, NP, Bb * C * NP);
}
